// LlamaWithSparseAttention_29712583754355
// MI455X (gfx1250) — matrix operand order repaired and checked by reading, not compiled, not run on hardware
//
#include <hip/hip_runtime.h>
#include <hip/hip_bf16.h>
#include <cstdint>
#include <cstddef>

// ---------------- problem constants ----------------
#define S_LEN   2048
#define DIMM    2048
#define NHEADS  32
#define KVH     8
#define DH      64
#define GF      4           // HEADS / KV_HEADS
#define CBLK    16
#define NSELK   8
#define WINL    64
#define CDIMM   1024        // CBLK*DH
#define NC      128         // S/CBLK
#define NJ      129         // NC + 1
#define NKEYF   144         // (NSEL+1)*SBLK
#define SCALE_F 0.125f      // DH^-0.5
#define NEGV    (-1e30f)
#define QKV_N   3072
#define K_COFF  2048
#define V_COFF  2560

typedef _Float16 h8  __attribute__((ext_vector_type(8)));
typedef _Float16 h16 __attribute__((ext_vector_type(16)));
typedef float    f8  __attribute__((ext_vector_type(8)));
typedef int      v4i __attribute__((vector_size(16)));   // GCC-style, matches builtin

// CDNA5 async global->LDS DMA path (ASYNCcnt), if the toolchain exposes it.
#if __has_builtin(__builtin_amdgcn_global_load_async_to_lds_b128) && \
    __has_builtin(__builtin_amdgcn_s_wait_asynccnt)
#define ASYNC_LDS 1
#define GPTR(p) ((__attribute__((address_space(1))) v4i*)(p))
#define LPTR(p) ((__attribute__((address_space(3))) v4i*)(p))
#else
#define ASYNC_LDS 0
#endif

// ---------------- fp32 -> fp16 convert ----------------
__global__ void f32_to_f16_kernel(const float* __restrict__ src,
                                  _Float16* __restrict__ dst, size_t n) {
  size_t id = (size_t)blockIdx.x * blockDim.x + threadIdx.x;
  if (id < n) dst[id] = (_Float16)src[id];
}

// ---------------- WMMA GEMM: C = A(MxK) @ B(KxN) [+bias][relu] ----------------
// A,B f16 row-major. Tile: BM=64 x BN, K-step 64, double-buffered LDS.
// 2x(BN/32) waves, each wave 32x32 out -> 8 v_wmma per wave per K-step.
template <int BN>
__global__ void gemm_wmma_f16(const _Float16* __restrict__ A,
                              const _Float16* __restrict__ B,
                              const float* __restrict__ bias,
                              float* __restrict__ Cf,
                              _Float16* __restrict__ Ch,
                              int M, int N, int K, int relu) {
  constexpr int BM   = 64;
  constexpr int KS   = 64;
  constexpr int WN   = BN / 32;
  constexpr int NTHR = 32 * 2 * WN;       // 256 (BN=128) / 128 (BN=64)
  constexpr int ACH  = (BM * KS / 8) / NTHR;  // A 16B-chunks per thread
  constexpr int BCH  = (KS * BN / 8) / NTHR;  // B 16B-chunks per thread (=4)

  __shared__ _Float16 Asl[2][BM * KS];    // [buf][row][k] row-major
  __shared__ _Float16 Bsl[2][BN * KS];    // [buf][n][k]  TRANSPOSED

  const int tid   = threadIdx.x;
  const int lane  = tid & 31;
  const int wave  = tid >> 5;
  const int waveM = wave / WN;
  const int waveN = wave % WN;
  const int lm    = lane & 15;
  const int lh    = lane >> 4;
  const int rowBase = blockIdx.y * BM;
  const int colBase = blockIdx.x * BN;
  const int nIter = K / KS;

  f8 acc[2][2] = {};
  h8 bReg[BCH];
#if !ASYNC_LDS
  h8 aReg[ACH];
#endif

  // --- staging helpers (all loops are wave-uniform: no exec masking) ---
  auto stageA_issue = [&](int it, int buf) {
    int k0 = it * KS;
#pragma unroll
    for (int u = 0; u < ACH; ++u) {
      int c = tid + u * NTHR;             // chunk id over 64x64/8
      int r = c >> 3, col = (c & 7) * 8;
#if ASYNC_LDS
      __builtin_amdgcn_global_load_async_to_lds_b128(
          GPTR(&A[(size_t)(rowBase + r) * K + k0 + col]),
          LPTR(&Asl[buf][r * KS + col]), 0, 0);
#else
      aReg[u] = *(const h8*)&A[(size_t)(rowBase + r) * K + k0 + col];
#endif
    }
  };
#if !ASYNC_LDS
  auto stageA_store = [&](int buf) {
#pragma unroll
    for (int u = 0; u < ACH; ++u) {
      int c = tid + u * NTHR;
      int r = c >> 3, col = (c & 7) * 8;
      *(h8*)&Asl[buf][r * KS + col] = aReg[u];
    }
  };
#endif
  auto stageB_load = [&](int it) {
    int k0 = it * KS;
#pragma unroll
    for (int u = 0; u < BCH; ++u) {
      int c = tid + u * NTHR;
      int kr = c / (BN / 8), nb = (c % (BN / 8)) * 8;
      bReg[u] = *(const h8*)&B[(size_t)(k0 + kr) * N + colBase + nb];
    }
  };
  auto stageB_store = [&](int buf) {
#pragma unroll
    for (int u = 0; u < BCH; ++u) {
      int c = tid + u * NTHR;
      int kr = c / (BN / 8), nb = (c % (BN / 8)) * 8;
#pragma unroll
      for (int e = 0; e < 8; ++e) Bsl[buf][(nb + e) * KS + kr] = bReg[u][e];
    }
  };
  auto compute = [&](int buf) {
#pragma unroll
    for (int ks = 0; ks < KS; ks += 32) {
      h16 af[2], bf[2];
#pragma unroll
      for (int mi = 0; mi < 2; ++mi) {
        // A frag (16x32): lane lm = row; halves 0-7 K=lh*8.., 8-15 K=16+lh*8..
        int r = waveM * 32 + mi * 16 + lm;
        h8 lo = *(const h8*)&Asl[buf][r * KS + ks + lh * 8];
        h8 hi = *(const h8*)&Asl[buf][r * KS + ks + 16 + lh * 8];
        af[mi] = __builtin_shufflevector(lo, hi, 0, 1, 2, 3, 4, 5, 6, 7,
                                         8, 9, 10, 11, 12, 13, 14, 15);
      }
#pragma unroll
      for (int ni = 0; ni < 2; ++ni) {
        // B frag (32x16): lane lm = col n; halves 0-7 K=lh*8.., 8-15 K=16+lh*8..
        int n = waveN * 32 + ni * 16 + lm;
        h8 lo = *(const h8*)&Bsl[buf][n * KS + ks + lh * 8];
        h8 hi = *(const h8*)&Bsl[buf][n * KS + ks + 16 + lh * 8];
        bf[ni] = __builtin_shufflevector(lo, hi, 0, 1, 2, 3, 4, 5, 6, 7,
                                         8, 9, 10, 11, 12, 13, 14, 15);
      }
#pragma unroll
      for (int mi = 0; mi < 2; ++mi)
#pragma unroll
        for (int ni = 0; ni < 2; ++ni)
          acc[mi][ni] = __builtin_amdgcn_wmma_f32_16x16x32_f16(
              false, af[mi], false, bf[ni], (short)0, acc[mi][ni], false,
              false);
    }
  };

  // --- pipeline: prologue ---
  stageA_issue(0, 0);
  stageB_load(0);
#if !ASYNC_LDS
  stageA_store(0);
#endif
  stageB_store(0);
#if ASYNC_LDS
  __builtin_amdgcn_s_wait_asynccnt(0);
#endif
  __syncthreads();

  for (int it = 0; it < nIter; ++it) {
    const int cur = it & 1, nxt = cur ^ 1;
    const bool more = (it + 1) < nIter;
    if (more) {
      stageA_issue(it + 1, nxt);          // async: DMA into nxt during compute
      stageB_load(it + 1);                // global loads in flight over WMMAs
    }
    if (it + 2 < nIter) {                 // prime L2 two tiles ahead
      int k0 = (it + 2) * KS;
      __builtin_prefetch(&A[(size_t)(rowBase + (tid >> 3)) * K + k0], 0, 1);
      __builtin_prefetch(&B[(size_t)(k0 + tid / (BN / 8)) * N + colBase], 0, 1);
    }
    compute(cur);
    if (more) {
#if !ASYNC_LDS
      stageA_store(nxt);
#endif
      stageB_store(nxt);
#if ASYNC_LDS
      __builtin_amdgcn_s_wait_asynccnt(0);
#endif
    }
    __syncthreads();
  }

  // epilogue: C/D layout: VGPR r -> row lh*8+r, col = lm
#pragma unroll
  for (int mi = 0; mi < 2; ++mi)
#pragma unroll
    for (int ni = 0; ni < 2; ++ni) {
      int gn = colBase + waveN * 32 + ni * 16 + lm;
      float bv = bias ? bias[gn] : 0.f;
#pragma unroll
      for (int r = 0; r < 8; ++r) {
        int gm = rowBase + waveM * 32 + mi * 16 + lh * 8 + r;
        float v = acc[mi][ni][r] + bv;
        if (relu) v = v > 0.f ? v : 0.f;
        size_t o = (size_t)gm * N + gn;
        if (Cf) Cf[o] = v;
        if (Ch) Ch[o] = (_Float16)v;
      }
    }
}

// ---------------- kb/vb build: blocks + positional, -> f16 rows (h*128+c, t*64+d)
__global__ void build_kv_blocks(const float* __restrict__ qkv,
                                const float* __restrict__ kpos,
                                const float* __restrict__ vpos,
                                _Float16* __restrict__ kbh,
                                _Float16* __restrict__ vbh) {
  size_t id = (size_t)blockIdx.x * 256 + threadIdx.x;
  if (id >= (size_t)KVH * NC * CDIMM) return;
  int idx = id & (CDIMM - 1);
  int r = (int)(id >> 10);
  int h = r >> 7, c = r & 127;
  int t = idx >> 6, d = idx & 63;
  size_t tok = (size_t)(c * CBLK + t) * QKV_N;
  kbh[id] = (_Float16)(qkv[tok + K_COFF + h * DH + d] +
                       kpos[((size_t)h * CBLK + t) * DH + d]);
  vbh[id] = (_Float16)(qkv[tok + V_COFF + h * DH + d] +
                       vpos[((size_t)h * CBLK + t) * DH + d]);
}

// ---------------- assemble ck/cv: [h][0]=mem, [h][1+c]=compressed
__global__ void assemble_c_kernel(const float* __restrict__ cktmp,
                                  const float* __restrict__ cvtmp,
                                  const float* __restrict__ memck,
                                  const float* __restrict__ memcv,
                                  float* __restrict__ ck, float* __restrict__ cv) {
  int id = blockIdx.x * 256 + threadIdx.x;
  if (id >= KVH * NJ * DH) return;
  int d = id & 63;
  int rest = id >> 6;
  int j = rest % NJ, h = rest / NJ;
  if (j == 0) {
    ck[id] = memck[h * DH + d];
    cv[id] = memcv[h * DH + d];
  } else {
    size_t s = ((size_t)h * NC + (j - 1)) * DH + d;
    ck[id] = cktmp[s];
    cv[id] = cvtmp[s];
  }
}

// ---------------- RoPE for q (per g) and k ----------------
__global__ void rope_kernel(const float* __restrict__ qkv,
                            float* __restrict__ rq, float* __restrict__ rk) {
  int id = blockIdx.x * 256 + threadIdx.x;   // over KVH*S*32 pairs
  if (id >= KVH * S_LEN * 32) return;
  int p = id & 31;
  int rest = id >> 5;
  int i = rest & (S_LEN - 1);
  int h = rest >> 11;
  float inv = __powf(10000.f, -(float)p / 32.f);
  float ang = (float)i * inv;
  float cs = __cosf(ang), sn = __sinf(ang);
  size_t kb = (size_t)i * QKV_N + K_COFF + h * DH + 2 * p;
  float x1 = qkv[kb], x2 = qkv[kb + 1];
  size_t ro = ((size_t)h * S_LEN + i) * DH + 2 * p;
  rk[ro] = x1 * cs - x2 * sn;
  rk[ro + 1] = x1 * sn + x2 * cs;
#pragma unroll
  for (int g = 0; g < GF; ++g) {
    size_t qb = (size_t)i * QKV_N + (h * GF + g) * DH + 2 * p;
    float y1 = qkv[qb], y2 = qkv[qb + 1];
    size_t qo = (((size_t)(h * GF + g)) * S_LEN + i) * DH + 2 * p;
    rq[qo] = y1 * cs - y2 * sn;
    rq[qo + 1] = y1 * sn + y2 * cs;
  }
}

// ---------------- compressed attention + importance ----------------
__global__ void cattn_kernel(const float* __restrict__ qkv,
                             const float* __restrict__ ck,
                             const float* __restrict__ cv,
                             float* __restrict__ cout, float* __restrict__ imp) {
  int blk = blockIdx.x;               // h*2048 + i
  int h = blk >> 11, i = blk & (S_LEN - 1);
  int tid = threadIdx.x;              // 128
  __shared__ float qs[GF][DH];
  __shared__ float s[GF][NJ];
  __shared__ float im[NJ];
  __shared__ float mg[GF], zg[GF], m2z2[2];

  for (int o = tid; o < GF * DH; o += 128) {
    int g = o >> 6, d = o & 63;
    qs[g][d] = qkv[(size_t)i * QKV_N + (h * GF + g) * DH + d];
  }
  __syncthreads();

  for (int j = tid; j < NJ; j += 128) {
    bool valid = (j == 0) || (j * CBLK <= i);
    const float* ckr = &ck[((size_t)h * NJ + j) * DH];
    for (int g = 0; g < GF; ++g) {
      float acc = NEGV;
      if (valid) {
        acc = 0.f;
        for (int d = 0; d < DH; ++d) acc += qs[g][d] * ckr[d];
        acc *= SCALE_F;
      }
      s[g][j] = acc;
    }
  }
  __syncthreads();

  if (tid < GF) {
    int g = tid;
    float m = -3.4e38f;
    for (int j = 0; j < NJ; ++j) m = fmaxf(m, s[g][j]);
    float z = 0.f;
    for (int j = 0; j < NJ; ++j) z += __expf(s[g][j] - m);
    mg[g] = m;
    zg[g] = z;
  }
  for (int j = tid; j < NJ; j += 128)
    im[j] = (j == 0) ? -1000.f
                     : 0.25f * (s[0][j] + s[1][j] + s[2][j] + s[3][j]);
  __syncthreads();
  if (tid == 0) {
    float m = -3.4e38f;
    for (int j = 0; j < NJ; ++j) m = fmaxf(m, im[j]);
    float z = 0.f;
    for (int j = 0; j < NJ; ++j) z += __expf(im[j] - m);
    m2z2[0] = m;
    m2z2[1] = z;
  }
  __syncthreads();

  // importance softmax probs for blocks 1..128
  for (int j = 1 + tid; j < NJ; j += 128)
    imp[((size_t)h * S_LEN + i) * NC + (j - 1)] =
        __expf(im[j] - m2z2[0]) / m2z2[1];

  // c_out
  for (int o = tid; o < GF * DH; o += 128) {
    int g = o >> 6, d = o & 63;
    float m = mg[g], z = zg[g], acc = 0.f;
    for (int j = 0; j < NJ; ++j) {
      float e = __expf(s[g][j] - m);
      acc += e * cv[((size_t)h * NJ + j) * DH + d];
    }
    cout[(((size_t)(h * GF + g)) * S_LEN + i) * DH + d] = acc / z;
  }
}

// ---------------- top-k (NSEL=8) per (h,i), tie -> lowest index ----------------
__global__ void topk_kernel(const float* __restrict__ imp,
                            int* __restrict__ seli, float* __restrict__ selm) {
  int id = blockIdx.x * blockDim.x + threadIdx.x;
  if (id >= KVH * S_LEN) return;
  const float* p = &imp[(size_t)id * NC];
  unsigned used[4] = {0u, 0u, 0u, 0u};
  for (int t = 0; t < NSELK; ++t) {
    float best = -1.f;
    int bi = 0;
    for (int j = 0; j < NC; ++j) {
      if (used[j >> 5] & (1u << (j & 31))) continue;
      float v = p[j];
      if (v > best) { best = v; bi = j; }
    }
    used[bi >> 5] |= 1u << (bi & 31);
    seli[(size_t)id * NSELK + t] = bi;
    selm[(size_t)id * NSELK + t] = (best > 1e-10f) ? 1.f : 0.f;
  }
}

// ---------------- fine attention: 8 selected + own block (144 keys) ----------------
__global__ void fattn_kernel(const float* __restrict__ qkv,
                             const float* __restrict__ rq,
                             const float* __restrict__ rk,
                             const int* __restrict__ seli,
                             const float* __restrict__ selm,
                             float* __restrict__ fout) {
  int blk = blockIdx.x;
  int h = blk >> 11, i = blk & (S_LEN - 1);
  int tid = threadIdx.x;              // 256
  __shared__ float qs[GF][DH];
  __shared__ float s[GF][NKEYF];
  __shared__ int tok[NKEYF];
  __shared__ float mg[GF], zg[GF];

  if (tid < GF * DH) {
    int g = tid >> 6, d = tid & 63;
    qs[g][d] = rq[(((size_t)(h * GF + g)) * S_LEN + i) * DH + d];
  }
  if (tid < NKEYF) {
    int b = (tid < NSELK * CBLK)
                ? seli[((size_t)h * S_LEN + i) * NSELK + (tid >> 4)]
                : (i >> 4);
    tok[tid] = b * CBLK + (tid & 15);
  }
  __syncthreads();

  if (tid < NKEYF) {
    bool valid =
        (tid < NSELK * CBLK)
            ? (selm[((size_t)h * S_LEN + i) * NSELK + (tid >> 4)] > 0.5f)
            : ((i & 15) >= (tid & 15));
    const float* kr = &rk[((size_t)h * S_LEN + tok[tid]) * DH];
    for (int g = 0; g < GF; ++g) {
      float acc = NEGV;
      if (valid) {
        acc = 0.f;
        for (int d = 0; d < DH; ++d) acc += qs[g][d] * kr[d];
        acc *= SCALE_F;
      }
      s[g][tid] = acc;
    }
  }
  __syncthreads();
  if (tid < GF) {
    int g = tid;
    float m = -3.4e38f;
    for (int j = 0; j < NKEYF; ++j) m = fmaxf(m, s[g][j]);
    float z = 0.f;
    for (int j = 0; j < NKEYF; ++j) z += __expf(s[g][j] - m);
    mg[g] = m;
    zg[g] = z;
  }
  __syncthreads();

  int g = tid >> 6, d = tid & 63;
  float m = mg[g], z = zg[g], acc = 0.f;
  for (int j = 0; j < NKEYF; ++j) {
    float e = __expf(s[g][j] - m);
    if (e != 0.f)
      acc += e * qkv[(size_t)tok[j] * QKV_N + V_COFF + h * DH + d];
  }
  fout[(((size_t)(h * GF + g)) * S_LEN + i) * DH + d] = acc / z;
}

// ---------------- sliding window attention (prev + current window, 128 keys) ----
__global__ void swin_kernel(const float* __restrict__ qkv,
                            const float* __restrict__ rq,
                            const float* __restrict__ rk,
                            float* __restrict__ sout) {
  int blk = blockIdx.x;
  int h = blk >> 11, i = blk & (S_LEN - 1);
  int w = i >> 6, il = i & 63;
  int tid = threadIdx.x;              // 128
  __shared__ float qs[GF][DH];
  __shared__ float s[GF][2 * WINL];
  __shared__ float mg[GF], zg[GF];

  for (int o = tid; o < GF * DH; o += 128) {
    int g = o >> 6, d = o & 63;
    qs[g][d] = rq[(((size_t)(h * GF + g)) * S_LEN + i) * DH + d];
  }
  __syncthreads();

  {
    int j = tid;
    int t = (w - 1) * WINL + j;
    bool valid = (j >= il) && (j <= il + WINL) && ((w > 0) || (j >= WINL));
    for (int g = 0; g < GF; ++g) {
      float acc = NEGV;
      if (valid) {
        const float* kr = &rk[((size_t)h * S_LEN + t) * DH];
        acc = 0.f;
        for (int d = 0; d < DH; ++d) acc += qs[g][d] * kr[d];
        acc *= SCALE_F;
      }
      s[g][j] = acc;
    }
  }
  __syncthreads();
  if (tid < GF) {
    int g = tid;
    float m = -3.4e38f;
    for (int j = 0; j < 2 * WINL; ++j) m = fmaxf(m, s[g][j]);
    float z = 0.f;
    for (int j = 0; j < 2 * WINL; ++j) z += __expf(s[g][j] - m);
    mg[g] = m;
    zg[g] = z;
  }
  __syncthreads();

  for (int o = tid; o < GF * DH; o += 128) {
    int g = o >> 6, d = o & 63;
    float m = mg[g], z = zg[g], acc = 0.f;
    for (int j = 0; j < 2 * WINL; ++j) {
      float e = __expf(s[g][j] - m);
      int t = (w - 1) * WINL + j;
      if (e != 0.f && t >= 0)
        acc += e * qkv[(size_t)t * QKV_N + V_COFF + h * DH + d];
    }
    sout[(((size_t)(h * GF + g)) * S_LEN + i) * DH + d] = acc / z;
  }
}

// ---------------- gates: sigmoid(x @ w_gate + b_gate) ----------------
__global__ void gates_kernel(const float* __restrict__ x,
                             const float* __restrict__ wg,
                             const float* __restrict__ bg,
                             float* __restrict__ gl) {
  int i = blockIdx.x;
  int c = threadIdx.x;                // 96
  __shared__ float xs[DIMM];
  for (int d = c; d < DIMM; d += 96) xs[d] = x[(size_t)i * DIMM + d];
  __syncthreads();
  float acc = bg[c];
  for (int d = 0; d < DIMM; ++d) acc += xs[d] * wg[(size_t)d * 96 + c];
  gl[(size_t)i * 96 + c] = 1.f / (1.f + __expf(-acc));
}

// ---------------- gated combine -> f16 rows for the final GEMM ----------------
__global__ void combine_kernel(const float* __restrict__ co,
                               const float* __restrict__ fo,
                               const float* __restrict__ so,
                               const float* __restrict__ gl,
                               _Float16* __restrict__ opre) {
  size_t id = (size_t)blockIdx.x * 256 + threadIdx.x;
  if (id >= (size_t)S_LEN * DIMM) return;
  int i = (int)(id >> 11);
  int col = (int)(id & (DIMM - 1));
  int hh = col >> 6, d = col & 63;
  size_t a = ((size_t)hh * S_LEN + i) * DH + d;
  const float* gp = &gl[(size_t)i * 96 + hh * 3];
  opre[id] = (_Float16)(gp[0] * co[a] + gp[1] * fo[a] + gp[2] * so[a]);
}

// ---------------- workspace carving ----------------
struct WsAlloc {
  char* base;
  size_t off;
  template <typename T>
  T* take(size_t count) {
    T* p = (T*)(base + off);
    off = (off + count * sizeof(T) + 255) & ~(size_t)255;
    return p;
  }
};

extern "C" void kernel_launch(void* const* d_in, const int* in_sizes, int n_in,
                              void* d_out, int out_size, void* d_ws,
                              size_t ws_size, hipStream_t stream) {
  (void)in_sizes; (void)n_in; (void)out_size; (void)ws_size;
  const float* x     = (const float*)d_in[0];
  const float* wqkv  = (const float*)d_in[1];
  const float* kpos  = (const float*)d_in[2];
  const float* vpos  = (const float*)d_in[3];
  const float* kcw1  = (const float*)d_in[4];
  const float* kcb1  = (const float*)d_in[5];
  const float* kcw2  = (const float*)d_in[6];
  const float* kcb2  = (const float*)d_in[7];
  const float* vcw1  = (const float*)d_in[8];
  const float* vcb1  = (const float*)d_in[9];
  const float* vcw2  = (const float*)d_in[10];
  const float* vcb2  = (const float*)d_in[11];
  const float* memck = (const float*)d_in[12];
  const float* memcv = (const float*)d_in[13];
  const float* wg    = (const float*)d_in[14];
  const float* bg    = (const float*)d_in[15];
  const float* wout  = (const float*)d_in[16];
  float* out = (float*)d_out;

  WsAlloc ws{(char*)d_ws, 0};
  _Float16* x_h    = ws.take<_Float16>((size_t)S_LEN * DIMM);
  _Float16* wqkv_h = ws.take<_Float16>((size_t)DIMM * QKV_N);
  float*    qkv    = ws.take<float>((size_t)S_LEN * QKV_N);
  _Float16* kcw1_h = ws.take<_Float16>((size_t)CDIMM * CDIMM);
  _Float16* vcw1_h = ws.take<_Float16>((size_t)CDIMM * CDIMM);
  _Float16* kcw2_h = ws.take<_Float16>((size_t)CDIMM * DH);
  _Float16* vcw2_h = ws.take<_Float16>((size_t)CDIMM * DH);
  _Float16* wout_h = ws.take<_Float16>((size_t)DIMM * DIMM);
  _Float16* kb_h   = ws.take<_Float16>((size_t)KVH * NC * CDIMM);
  _Float16* vb_h   = ws.take<_Float16>((size_t)KVH * NC * CDIMM);
  _Float16* h1k_h  = ws.take<_Float16>((size_t)KVH * NC * CDIMM);
  _Float16* h1v_h  = ws.take<_Float16>((size_t)KVH * NC * CDIMM);
  float*    cktmp  = ws.take<float>((size_t)KVH * NC * DH);
  float*    cvtmp  = ws.take<float>((size_t)KVH * NC * DH);
  float*    ck     = ws.take<float>((size_t)KVH * NJ * DH);
  float*    cv     = ws.take<float>((size_t)KVH * NJ * DH);
  float*    rq     = ws.take<float>((size_t)NHEADS * S_LEN * DH);
  float*    rk     = ws.take<float>((size_t)KVH * S_LEN * DH);
  float*    co     = ws.take<float>((size_t)NHEADS * S_LEN * DH);
  float*    fo     = ws.take<float>((size_t)NHEADS * S_LEN * DH);
  float*    so     = ws.take<float>((size_t)NHEADS * S_LEN * DH);
  float*    imp    = ws.take<float>((size_t)KVH * S_LEN * NC);
  int*      seli   = ws.take<int>((size_t)KVH * S_LEN * NSELK);
  float*    selm   = ws.take<float>((size_t)KVH * S_LEN * NSELK);
  float*    gl     = ws.take<float>((size_t)S_LEN * 96);
  _Float16* opre_h = ws.take<_Float16>((size_t)S_LEN * DIMM);

  auto conv = [&](const float* s, _Float16* d, size_t n) {
    f32_to_f16_kernel<<<(unsigned)((n + 255) / 256), 256, 0, stream>>>(s, d, n);
  };
  conv(x,    x_h,    (size_t)S_LEN * DIMM);
  conv(wqkv, wqkv_h, (size_t)DIMM * QKV_N);
  conv(kcw1, kcw1_h, (size_t)CDIMM * CDIMM);
  conv(vcw1, vcw1_h, (size_t)CDIMM * CDIMM);
  conv(kcw2, kcw2_h, (size_t)CDIMM * DH);
  conv(vcw2, vcw2_h, (size_t)CDIMM * DH);
  conv(wout, wout_h, (size_t)DIMM * DIMM);

  // qkv = x @ w_qkv  (2048 x 2048 x 3072)
  gemm_wmma_f16<128><<<dim3(QKV_N / 128, S_LEN / 64), 256, 0, stream>>>(
      x_h, wqkv_h, nullptr, qkv, nullptr, S_LEN, QKV_N, DIMM, 0);

  // kb/vb blocks (+positional) -> f16
  build_kv_blocks<<<(KVH * NC * CDIMM) / 256, 256, 0, stream>>>(
      qkv, kpos, vpos, kb_h, vb_h);

  // compression MLPs: h1 = relu(kb@W1+b1) (f16 out), ck = h1@W2+b2 (f32)
  gemm_wmma_f16<128><<<dim3(CDIMM / 128, (KVH * NC) / 64), 256, 0, stream>>>(
      kb_h, kcw1_h, kcb1, nullptr, h1k_h, KVH * NC, CDIMM, CDIMM, 1);
  gemm_wmma_f16<64><<<dim3(1, (KVH * NC) / 64), 128, 0, stream>>>(
      h1k_h, kcw2_h, kcb2, cktmp, nullptr, KVH * NC, DH, CDIMM, 0);
  gemm_wmma_f16<128><<<dim3(CDIMM / 128, (KVH * NC) / 64), 256, 0, stream>>>(
      vb_h, vcw1_h, vcb1, nullptr, h1v_h, KVH * NC, CDIMM, CDIMM, 1);
  gemm_wmma_f16<64><<<dim3(1, (KVH * NC) / 64), 128, 0, stream>>>(
      h1v_h, vcw2_h, vcb2, cvtmp, nullptr, KVH * NC, DH, CDIMM, 0);

  assemble_c_kernel<<<(KVH * NJ * DH + 255) / 256, 256, 0, stream>>>(
      cktmp, cvtmp, memck, memcv, ck, cv);

  rope_kernel<<<(KVH * S_LEN * 32) / 256, 256, 0, stream>>>(qkv, rq, rk);

  cattn_kernel<<<KVH * S_LEN, 128, 0, stream>>>(qkv, ck, cv, co, imp);
  topk_kernel<<<(KVH * S_LEN) / 256, 256, 0, stream>>>(imp, seli, selm);
  fattn_kernel<<<KVH * S_LEN, 256, 0, stream>>>(qkv, rq, rk, seli, selm, fo);
  swin_kernel<<<KVH * S_LEN, 128, 0, stream>>>(qkv, rq, rk, so);

  gates_kernel<<<S_LEN, 96, 0, stream>>>(x, wg, bg, gl);
  combine_kernel<<<((size_t)S_LEN * DIMM) / 256, 256, 0, stream>>>(
      co, fo, so, gl, opre_h);

  // out = out_pre @ w_out  (2048 x 2048 x 2048)
  gemm_wmma_f16<128><<<dim3(DIMM / 128, S_LEN / 64), 256, 0, stream>>>(
      opre_h, wout_h, nullptr, out, nullptr, S_LEN, DIMM, DIMM, 0);
}
